// NonLocalVisualEncoder_59279138619579
// MI455X (gfx1250) — matrix operand order repaired and checked by reading, not compiled, not run on hardware
//
#include <hip/hip_runtime.h>
#include <cstdint>
#include <cstddef>

// ---------- raw vector types (trivially-copyable, safe in unions) ----------
typedef __attribute__((ext_vector_type(16))) __bf16       bf16x16;
typedef __attribute__((ext_vector_type(8)))  float        f32x8;
typedef __attribute__((ext_vector_type(4)))  float        f32x4;
typedef __attribute__((ext_vector_type(4)))  unsigned int u32x4;
typedef __attribute__((ext_vector_type(2)))  unsigned int u32x2;

union Frag { bf16x16 v; u32x4 q[2]; };   // 32 bytes: one WMMA A/B operand

__device__ __forceinline__ unsigned int f2bf(float f) {
  unsigned int u = __float_as_uint(f);
  u += 0x7FFFu + ((u >> 16) & 1u);       // round-to-nearest-even
  return u >> 16;
}
__device__ __forceinline__ unsigned int pk2(float a, float b) {
  return f2bf(a) | (f2bf(b) << 16);
}

// gfx1250 async copy: global -> LDS, 16 bytes per lane, tracked by ASYNCcnt.
__device__ __forceinline__ void async_ld16(unsigned lds_off, const void* gptr) {
  asm volatile("global_load_async_to_lds_b128 %0, %1, off"
               :: "v"(lds_off), "v"((unsigned long long)gptr)
               : "memory");
}
__device__ __forceinline__ void wait_async0() {
  asm volatile("s_wait_asynccnt 0x0" ::: "memory");
}

// ---------------- tiling ----------------
constexpr int BM = 128, BN = 128, KC = 64;
constexpr int LDSK = KC + 8;             // 144B row stride (16B aligned, bank spread)

// One WMMA GEMM: C[m,n] = sum_k A[m,k] * Bt[n,k]  (+bias[n]) (+resid[m,n])
// AF32/BF32: source is f32 (convert to bf16 while staging) else bf16 via async-to-LDS.
// OBF16: store bf16 (ushort) else f32.   OT: store transposed C[n*ldc + m].
template <int AF32, int BF32, int OBF16, int OT>
__global__ __launch_bounds__(256) void wmma_gemm(
    const void* __restrict__ Ap, long long lda, long long strideAb,
    const void* __restrict__ Btp, long long ldb, long long strideBb,
    const float* __restrict__ bias,
    const float* __restrict__ resid, long long ldr,
    void* __restrict__ Cp, long long ldc, long long strideCb,
    int M, int Nd, int K) {
  __shared__ unsigned short As[BM][LDSK];
  __shared__ unsigned short Bs[BN][LDSK];

  const int tid  = threadIdx.x;
  const int lane = tid & 31;             // wave32 lane
  const int wave = tid >> 5;             // 8 waves
  const int wr   = wave >> 2;            // 0..1 : 64-row group
  const int wc   = wave & 3;             // 0..3 : 32-col group
  const int half = lane >> 4;
  const int l16  = lane & 15;

  const long long m0 = (long long)blockIdx.y * BM;
  const long long n0 = (long long)blockIdx.x * BN;
  const int bb = blockIdx.z;

  const int lrow = tid >> 1;             // 0..127 : tile row this thread stages
  const int lseg = (tid & 1) * 32;       // 0 / 32 : K half of the 64-wide chunk

  const float*          Af = (const float*)Ap + (long long)bb * strideAb;
  const unsigned short* Ah = (const unsigned short*)Ap + (long long)bb * strideAb;
  const float*          Bf = (const float*)Btp + (long long)bb * strideBb;
  const unsigned short* Bh = (const unsigned short*)Btp + (long long)bb * strideBb;

  f32x8 acc[4][2];
  {
    f32x8 z = {0.f, 0.f, 0.f, 0.f, 0.f, 0.f, 0.f, 0.f};
    for (int i = 0; i < 4; ++i)
      for (int j = 0; j < 2; ++j) acc[i][j] = z;
  }

  const int ksteps = K / KC;
  for (int kk = 0; kk < ksteps; ++kk) {
    const int k0 = kk * KC;
    // ---- stage A tile (128 x 64 bf16) ----
    {
      if constexpr (AF32) {
        const float* s = Af + (m0 + lrow) * lda + k0 + lseg;
#pragma unroll
        for (int c = 0; c < 4; ++c) {
          f32x4 f0 = *(const f32x4*)(s + c * 8);
          f32x4 f1 = *(const f32x4*)(s + c * 8 + 4);
          u32x4 o = {pk2(f0.x, f0.y), pk2(f0.z, f0.w),
                     pk2(f1.x, f1.y), pk2(f1.z, f1.w)};
          *(u32x4*)&As[lrow][lseg + c * 8] = o;
        }
      } else {
        const unsigned short* s = Ah + (m0 + lrow) * lda + k0 + lseg;
#pragma unroll
        for (int c = 0; c < 4; ++c)
          async_ld16((unsigned)(unsigned long long)&As[lrow][lseg + c * 8],
                     s + c * 8);
      }
    }
    // ---- stage Bt tile (128 x 64 bf16) ----
    {
      if constexpr (BF32) {
        const float* s = Bf + (n0 + lrow) * ldb + k0 + lseg;
#pragma unroll
        for (int c = 0; c < 4; ++c) {
          f32x4 f0 = *(const f32x4*)(s + c * 8);
          f32x4 f1 = *(const f32x4*)(s + c * 8 + 4);
          u32x4 o = {pk2(f0.x, f0.y), pk2(f0.z, f0.w),
                     pk2(f1.x, f1.y), pk2(f1.z, f1.w)};
          *(u32x4*)&Bs[lrow][lseg + c * 8] = o;
        }
      } else {
        const unsigned short* s = Bh + (n0 + lrow) * ldb + k0 + lseg;
#pragma unroll
        for (int c = 0; c < 4; ++c)
          async_ld16((unsigned)(unsigned long long)&Bs[lrow][lseg + c * 8],
                     s + c * 8);
      }
    }
    // ---- prefetch next K chunk (global_prefetch_b8) ----
    if (kk + 1 < ksteps) {
      if constexpr (AF32) __builtin_prefetch(Af + (m0 + lrow) * lda + k0 + KC + lseg, 0, 0);
      else                __builtin_prefetch(Ah + (m0 + lrow) * lda + k0 + KC + lseg, 0, 0);
      if constexpr (BF32) __builtin_prefetch(Bf + (n0 + lrow) * ldb + k0 + KC + lseg, 0, 0);
      else                __builtin_prefetch(Bh + (n0 + lrow) * ldb + k0 + KC + lseg, 0, 0);
    }
    if constexpr (!(AF32 && BF32)) wait_async0();  // drain this wave's async-to-LDS
    __syncthreads();

    // ---- compute: 2 K-steps x 8 wmma per wave per stage ----
#pragma unroll
    for (int kc = 0; kc < 2; ++kc) {
      Frag bfr[2];
#pragma unroll
      for (int nt = 0; nt < 2; ++nt) {
        const int n  = wc * 32 + nt * 16 + l16;   // Bt row (= output col)
        const int kb = kc * 32 + half * 16;       // 16 contiguous K per lane
        bfr[nt].q[0] = *(const u32x4*)&Bs[n][kb];
        bfr[nt].q[1] = *(const u32x4*)&Bs[n][kb + 8];
      }
      Frag afr[4];
#pragma unroll
      for (int mt = 0; mt < 4; ++mt) {
        const int ar = wr * 64 + mt * 16 + l16;   // A row
        const int ka = kc * 32 + half * 16;       // 16 contiguous K per lane
        afr[mt].q[0] = *(const u32x4*)&As[ar][ka];
        afr[mt].q[1] = *(const u32x4*)&As[ar][ka + 8];
      }
#pragma unroll
      for (int mt = 0; mt < 4; ++mt)
#pragma unroll
        for (int nt = 0; nt < 2; ++nt)
          acc[mt][nt] = __builtin_amdgcn_wmma_f32_16x16x32_bf16(
              false, afr[mt].v, false, bfr[nt].v, (short)0, acc[mt][nt],
              false, false);
    }
    __syncthreads();
  }

  // ---- epilogue: C layout m = r + 8*half, n = lane&15 ----
#pragma unroll
  for (int mt = 0; mt < 4; ++mt) {
#pragma unroll
    for (int nt = 0; nt < 2; ++nt) {
      const long long n = n0 + wc * 32 + nt * 16 + l16;
      const float bv = bias ? bias[n] : 0.f;
      if constexpr (OT && OBF16) {
        // lane's 8 values are consecutive m -> contiguous in transposed layout
        const long long mb = m0 + wr * 64 + mt * 16 + half * 8;
        const f32x8 a = acc[mt][nt];
        u32x4 o = {pk2(a[0] + bv, a[1] + bv), pk2(a[2] + bv, a[3] + bv),
                   pk2(a[4] + bv, a[5] + bv), pk2(a[6] + bv, a[7] + bv)};
        *(u32x4*)&((unsigned short*)Cp)[(long long)bb * strideCb + n * ldc + mb] = o;
      } else {
#pragma unroll
        for (int r = 0; r < 8; ++r) {
          const long long m = m0 + wr * 64 + mt * 16 + half * 8 + r;
          float v = acc[mt][nt][r] + bv;
          if (resid) v += resid[m * ldr + n];
          const long long ci = (long long)bb * strideCb +
                               (OT ? (n * ldc + m) : (m * ldc + n));
          if constexpr (OBF16) ((unsigned short*)Cp)[ci] = (unsigned short)f2bf(v);
          else                 ((float*)Cp)[ci] = v;
        }
      }
    }
  }
}

// Row softmax over 4096 f32, writing normalized bf16 probs IN PLACE over the
// row's own storage (write bytes [0,8K) of the 16K row slot; all reads precede
// writes via the block reductions' barriers).
__global__ __launch_bounds__(256) void softmax_rows(const float* __restrict__ S,
                                                    unsigned short* __restrict__ P,
                                                    int ncols, long long prow) {
  __shared__ float red[256];
  const int tid = threadIdx.x;
  const long long row = blockIdx.x;
  const float* src = S + row * (long long)ncols;

  float v[16];
  float m = -3.402823466e38f;
#pragma unroll
  for (int c = 0; c < 4; ++c) {
    f32x4 f = *(const f32x4*)(src + c * 1024 + tid * 4);
    v[c * 4 + 0] = f.x; v[c * 4 + 1] = f.y; v[c * 4 + 2] = f.z; v[c * 4 + 3] = f.w;
    m = fmaxf(m, fmaxf(fmaxf(f.x, f.y), fmaxf(f.z, f.w)));
  }
  red[tid] = m; __syncthreads();
  for (int s = 128; s > 0; s >>= 1) {
    if (tid < s) red[tid] = fmaxf(red[tid], red[tid + s]);
    __syncthreads();
  }
  m = red[0]; __syncthreads();

  float sum = 0.f;
#pragma unroll
  for (int i = 0; i < 16; ++i) { v[i] = __expf(v[i] - m); sum += v[i]; }
  red[tid] = sum; __syncthreads();
  for (int s = 128; s > 0; s >>= 1) {
    if (tid < s) red[tid] += red[tid + s];
    __syncthreads();
  }
  const float inv = 1.f / red[0];

  unsigned short* dst = P + row * prow;
#pragma unroll
  for (int c = 0; c < 4; ++c) {
    u32x2 pk = {pk2(v[c * 4 + 0] * inv, v[c * 4 + 1] * inv),
                pk2(v[c * 4 + 2] * inv, v[c * 4 + 3] * inv)};
    *(u32x2*)(dst + c * 1024 + tid * 4) = pk;
  }
}

extern "C" void kernel_launch(void* const* d_in, const int* in_sizes, int n_in,
                              void* d_out, int out_size, void* d_ws, size_t ws_size,
                              hipStream_t stream) {
  (void)in_sizes; (void)n_in; (void)out_size; (void)ws_size;
  constexpr int B = 4, N = 4096, D = 2048, E = 1024;

  const float* x   = (const float*)d_in[0];
  const float* thw = (const float*)d_in[1];
  const float* thb = (const float*)d_in[2];
  const float* phw = (const float*)d_in[3];
  const float* phb = (const float*)d_in[4];
  const float* gw  = (const float*)d_in[5];
  const float* gb  = (const float*)d_in[6];
  const float* Ww  = (const float*)d_in[7];
  const float* Wb  = (const float*)d_in[8];

  // ---- workspace carve-up (~384 MB) ----
  char* ws = (char*)d_ws;
  size_t off = 0;
  auto carve = [&](size_t bytes) {
    char* p = ws + off;
    off += (bytes + 255) & ~(size_t)255;
    return p;
  };
  unsigned short* theta = (unsigned short*)carve((size_t)B * N * E * 2);
  unsigned short* phi   = (unsigned short*)carve((size_t)B * N * E * 2);
  unsigned short* gT    = (unsigned short*)carve((size_t)B * E * N * 2); // per-batch [E][N]
  unsigned short* ybuf  = (unsigned short*)carve((size_t)B * N * E * 2);
  float*          S     = (float*)carve((size_t)B * N * N * 4);
  unsigned short* P     = (unsigned short*)S;  // bf16 probs, row stride 2N elems

  const dim3 blk(256);

  // 1) theta = x @ thw^T + thb   (M=B*N, Nd=E, K=D)
  wmma_gemm<1, 1, 1, 0><<<dim3(E / BN, (B * N) / BM, 1), blk, 0, stream>>>(
      x, D, 0, thw, D, 0, thb, nullptr, 0, theta, E, 0, B * N, E, D);
  // 2) phi
  wmma_gemm<1, 1, 1, 0><<<dim3(E / BN, (B * N) / BM, 1), blk, 0, stream>>>(
      x, D, 0, phw, D, 0, phb, nullptr, 0, phi, E, 0, B * N, E, D);
  // 3) gT (transposed store, batched over B)
  wmma_gemm<1, 1, 1, 1><<<dim3(E / BN, N / BM, B), blk, 0, stream>>>(
      x, D, (long long)N * D, gw, D, 0, gb, nullptr, 0,
      gT, N, (long long)E * N, N, E, D);
  // 4) S = theta @ phi^T   (bf16 in via async-to-LDS, f32 out, batched)
  wmma_gemm<0, 0, 0, 0><<<dim3(N / BN, N / BM, B), blk, 0, stream>>>(
      theta, E, (long long)N * E, phi, E, (long long)N * E, nullptr, nullptr, 0,
      S, N, (long long)N * N, N, N, E);
  // 5) softmax rows -> bf16 probs in place
  softmax_rows<<<dim3(B * N), blk, 0, stream>>>(S, P, N, 2LL * N);
  // 6) y = P @ g   (A = P bf16, lda = 2N; Bt = gT)
  wmma_gemm<0, 0, 1, 0><<<dim3(E / BN, N / BM, B), blk, 0, stream>>>(
      P, 2LL * N, 2LL * (long long)N * N, gT, N, (long long)E * N,
      nullptr, nullptr, 0, ybuf, E, (long long)N * E, N, E, N);
  // 7) z = y @ Ww^T + Wb + x
  wmma_gemm<0, 1, 0, 0><<<dim3(D / BN, (B * N) / BM, 1), blk, 0, stream>>>(
      ybuf, E, 0, Ww, E, 0, Wb, x, D, d_out, D, 0, B * N, D, E);
}
